// Int4Linear_2997887172645
// MI455X (gfx1250) — matrix operand order repaired and checked by reading, not compiled, not run on hardware
//
#include <hip/hip_runtime.h>

typedef __attribute__((ext_vector_type(16))) _Float16 v16h;
typedef __attribute__((ext_vector_type(8)))  _Float16 v8h;
typedef __attribute__((ext_vector_type(2)))  _Float16 v2h;
typedef __attribute__((ext_vector_type(8)))  float    v8f;
typedef __attribute__((ext_vector_type(4)))  float    v4f;
typedef __attribute__((ext_vector_type(4)))  int      v4i;
typedef __attribute__((ext_vector_type(2)))  float    v2f;

#define M_TOK   64
#define K_FEAT  8192
#define N_FEAT  8192
#define GROUPSZ 128
#define NGROUPS (K_FEAT / GROUPSZ)   // 64
#define NSTEPS  (K_FEAT / 32)        // 256 WMMA K-steps
#define NB      64                   // output columns per block (4 n-waves x 16)

// ---------- pre-pass: convert activations fp32 -> f16 (row-major, same layout)
__global__ __launch_bounds__(256)
void cvt_a_kernel(const float* __restrict__ x, _Float16* __restrict__ y) {
    const int i = (blockIdx.x * 256 + threadIdx.x) * 8;
    const v4f a = *(const v4f*)(x + i);
    const v4f b = *(const v4f*)(x + i + 4);
    v8h h;
#pragma unroll
    for (int j = 0; j < 4; ++j) {
        h[j]     = (_Float16)a[j];
        h[4 + j] = (_Float16)b[j];
    }
    *(v8h*)(y + i) = h;
}

// dequant 2 codes -> half2 via exponent-bias trick:
// bitcast<half>(0x6400|q) == 1024+q  (exact for q in [0,1024))
// (q-8)*s+z == (1024+q)*s + (z-1032*s)
__device__ __forceinline__ v2h dq2(int q0, int q1, v2h s2, v2h c2) {
    unsigned p = ((unsigned)q0 | ((unsigned)q1 << 16)) | 0x64006400u;
    v2h h = __builtin_bit_cast(v2h, p);
    return h * s2 + c2;   // v_pk_fma_f16
}

// One pipeline stage: 16 weight codes + two A fragments (as 4 x v8h)
template <typename AT>
struct StepBuf {
    v4i q0, q1, q2, q3;
    v8h a0l, a0h, a1l, a1h;
};

template <typename AT>
__device__ __forceinline__ void load_step(StepBuf<AT>& sb,
                                          const int* __restrict__ wrow,
                                          const AT*  __restrict__ arow0,
                                          const AT*  __restrict__ arow1,
                                          int k0) {
    sb.q0 = *(const v4i*)(wrow + k0);
    sb.q1 = *(const v4i*)(wrow + k0 + 4);
    sb.q2 = *(const v4i*)(wrow + k0 + 16);
    sb.q3 = *(const v4i*)(wrow + k0 + 20);
    if constexpr (sizeof(AT) == 2) {
        sb.a0l = *(const v8h*)(arow0 + k0);
        sb.a0h = *(const v8h*)(arow0 + k0 + 16);
        sb.a1l = *(const v8h*)(arow1 + k0);
        sb.a1h = *(const v8h*)(arow1 + k0 + 16);
    } else {
        const v4f x00 = *(const v4f*)(arow0 + k0);
        const v4f x01 = *(const v4f*)(arow0 + k0 + 4);
        const v4f x02 = *(const v4f*)(arow0 + k0 + 16);
        const v4f x03 = *(const v4f*)(arow0 + k0 + 20);
        const v4f x10 = *(const v4f*)(arow1 + k0);
        const v4f x11 = *(const v4f*)(arow1 + k0 + 4);
        const v4f x12 = *(const v4f*)(arow1 + k0 + 16);
        const v4f x13 = *(const v4f*)(arow1 + k0 + 20);
#pragma unroll
        for (int j = 0; j < 4; ++j) {
            sb.a0l[j] = (_Float16)x00[j]; sb.a0l[4 + j] = (_Float16)x01[j];
            sb.a0h[j] = (_Float16)x02[j]; sb.a0h[4 + j] = (_Float16)x03[j];
            sb.a1l[j] = (_Float16)x10[j]; sb.a1l[4 + j] = (_Float16)x11[j];
            sb.a1h[j] = (_Float16)x12[j]; sb.a1h[4 + j] = (_Float16)x13[j];
        }
    }
}

template <typename AT>
__device__ __forceinline__ void step_mma(const StepBuf<AT>& sb, v2h s2, v2h c2,
                                         v8f& acc0, v8f& acc1) {
    const v2h w0 = dq2(sb.q0[0], sb.q0[1], s2, c2);
    const v2h w1 = dq2(sb.q0[2], sb.q0[3], s2, c2);
    const v2h w2 = dq2(sb.q1[0], sb.q1[1], s2, c2);
    const v2h w3 = dq2(sb.q1[2], sb.q1[3], s2, c2);
    const v2h w4 = dq2(sb.q2[0], sb.q2[1], s2, c2);
    const v2h w5 = dq2(sb.q2[2], sb.q2[3], s2, c2);
    const v2h w6 = dq2(sb.q3[0], sb.q3[1], s2, c2);
    const v2h w7 = dq2(sb.q3[2], sb.q3[3], s2, c2);
    v16h b;
#pragma unroll
    for (int j = 0; j < 2; ++j) {
        b[j]      = w0[j]; b[2 + j]  = w1[j];
        b[4 + j]  = w2[j]; b[6 + j]  = w3[j];
        b[8 + j]  = w4[j]; b[10 + j] = w5[j];
        b[12 + j] = w6[j]; b[14 + j] = w7[j];
    }
    v16h a0, a1;
#pragma unroll
    for (int j = 0; j < 8; ++j) {
        a0[j] = sb.a0l[j]; a0[8 + j] = sb.a0h[j];
        a1[j] = sb.a1l[j]; a1[8 + j] = sb.a1h[j];
    }
    acc0 = __builtin_amdgcn_wmma_f32_16x16x32_f16(false, a0, false, b, (short)0, acc0, false, false);
    acc1 = __builtin_amdgcn_wmma_f32_16x16x32_f16(false, a1, false, b, (short)0, acc1, false, false);
}

// One wave = 32x16 C tile (two 16x16 WMMA tiles sharing one dequantized B).
// 8 waves/block arranged 2 (M-half) x 4 (N): block covers M=64 x N=64.
// Grid.x = N_FEAT/64 = 128.  Software-pipelined 2 deep over K-steps.
template <typename AT>   // AT = _Float16 (pre-converted in ws) or float (fallback)
__global__ __launch_bounds__(256)
void int4linear_wmma_kernel(const AT*    __restrict__ inpA,    // [64, 8192]
                            const int*   __restrict__ wq,      // [8192, 8192] codes 0..15
                            const float* __restrict__ sz,      // [64, 8192, 2] (scale, zero)
                            const float* __restrict__ bias,    // [8192]
                            float*       __restrict__ out)     // [64, 8192]
{
    const int tid   = threadIdx.x;
    const int wave  = tid >> 5;         // 0..7
    const int lane  = tid & 31;
    const int lan16 = lane & 15;
    const int hi    = lane >> 4;        // K-half / interleave select

    const int m_half = (wave & 1) << 5;                         // 0 or 32
    const int n      = blockIdx.x * NB + ((wave >> 1) << 4) + lan16;

    const int* __restrict__ wrow = wq + (size_t)n * K_FEAT + (hi << 3);
    const float bn = bias[n];

    const AT* __restrict__ arow0 = inpA + (size_t)(m_half +      lan16) * K_FEAT + (hi << 3);
    const AT* __restrict__ arow1 = inpA + (size_t)(m_half + 16 + lan16) * K_FEAT + (hi << 3);

    v8f acc0 = {};
    v8f acc1 = {};

    StepBuf<AT> cur, nxt;
    load_step(cur, wrow, arow0, arow1, 0);

    for (int g = 0; g < NGROUPS; ++g) {
        const v2f szv = *(const v2f*)(sz + (((size_t)g * N_FEAT + n) << 1));
        const _Float16 sh = (_Float16)szv.x;
        const _Float16 ch = (_Float16)(szv.y - 1032.0f * szv.x);
        const v2h s2 = {sh, sh};
        const v2h c2 = {ch, ch};

        if (g + 1 < NGROUPS)
            __builtin_prefetch(wrow + (g + 1) * GROUPSZ, 0, 1);

#pragma unroll
        for (int kk = 0; kk < GROUPSZ / 32; ++kk) {
            const int ks = g * (GROUPSZ / 32) + kk;
            // issue next step's loads first (clamped at the end: redundant reload)
            const int kn = (ks + 1 < NSTEPS ? ks + 1 : ks) * 32;
            load_step(nxt, wrow, arow0, arow1, kn);
            // compute current step (dequant + 2 WMMAs)
            step_mma(cur, s2, c2, acc0, acc1);
            cur = nxt;
        }
    }

    float* __restrict__ orow0 = out + (size_t)(m_half +      (hi << 3)) * N_FEAT + n;
    float* __restrict__ orow1 = out + (size_t)(m_half + 16 + (hi << 3)) * N_FEAT + n;
#pragma unroll
    for (int r = 0; r < 8; ++r) {
        orow0[(size_t)r * N_FEAT] = acc0[r] + bn;
        orow1[(size_t)r * N_FEAT] = acc1[r] + bn;
    }
}

extern "C" void kernel_launch(void* const* d_in, const int* in_sizes, int n_in,
                              void* d_out, int out_size, void* d_ws, size_t ws_size,
                              hipStream_t stream) {
    const float* inp  = (const float*)d_in[0];
    const int*   wq   = (const int*)d_in[1];
    const float* sz   = (const float*)d_in[2];
    const float* bias = (const float*)d_in[3];
    float*       out  = (float*)d_out;

    const size_t aBytes = (size_t)M_TOK * K_FEAT * sizeof(_Float16);  // 1 MB
    dim3 grid(N_FEAT / NB);   // 128 blocks
    dim3 block(256);          // 8 waves (wave32)

    if (ws_size >= aBytes) {
        _Float16* a16 = (_Float16*)d_ws;
        cvt_a_kernel<<<dim3((M_TOK * K_FEAT) / (256 * 8)), 256, 0, stream>>>(inp, a16);
        int4linear_wmma_kernel<_Float16><<<grid, block, 0, stream>>>(a16, wq, sz, bias, out);
    } else {
        int4linear_wmma_kernel<float><<<grid, block, 0, stream>>>(inp, wq, sz, bias, out);
    }
}
